// ZonosAttention_35691178230293
// MI455X (gfx1250) — compile-verified, matrix operand order checked by reading, not run on hardware
//
#include <hip/hip_runtime.h>
#include <hip/hip_bf16.h>

typedef __bf16 bf16;
typedef __attribute__((ext_vector_type(16))) bf16  v16bf;
typedef __attribute__((ext_vector_type(8)))  bf16  v8bf;
typedef __attribute__((ext_vector_type(8)))  float v8f;

#define WMMA_BF16(A, B, C) \
  __builtin_amdgcn_wmma_f32_16x16x32_bf16(false, (A), false, (B), (short)0, (C), false, false)

// ---------------------------------------------------------------------------
// Fragment loader for v_wmma_f32_16x16x32_bf16 operands.
// A operand: lane<16 -> row=lane, K chunks {koff+0..7, koff+16..23}
//            lane>=16 -> row=lane-16, K chunks {koff+8..15, koff+24..31}
// B operand: identical with "row" = output column (matrix stored [N][K]).
// stride must be a multiple of 8 elements (16B-aligned b128 loads).
// stride 72 (144B rows): 16 lanes' b128 ops land on 16 distinct bank groups.
// ---------------------------------------------------------------------------
__device__ __forceinline__ v16bf frag_ld(const bf16* base, int row_base, int stride, int koff) {
  const int lane = threadIdx.x & 31;
  const int half = lane >> 4;
  const bf16* p = base + (size_t)(row_base + (lane & 15)) * stride + koff + half * 8;
  v8bf lo = *(const v8bf*)(p);
  v8bf hi = *(const v8bf*)(p + 16);
  v16bf f;
#pragma unroll
  for (int i = 0; i < 8; ++i) { f[i] = lo[i]; f[i + 8] = hi[i]; }
  return f;
}

__device__ __forceinline__ v8f v8f_zero() {
  v8f z;
#pragma unroll
  for (int i = 0; i < 8; ++i) z[i] = 0.0f;
  return z;
}

// ---------------------------------------------------------------------------
// Tensor Data Mover: 2D tile load Global->LDS with hardware row padding that
// reproduces the stride-72 bf16 LDS layout (pad_interval=32 dwords -> after
// every 64 bf16, pad_amount=4 dwords -> +8 bf16). Descriptor packing per
// CDNA5 ISA. Guarded: falls back to
// cooperative vector loads if the builtin is absent.
// ---------------------------------------------------------------------------
#if defined(__has_builtin)
#if __has_builtin(__builtin_amdgcn_tensor_load_to_lds)
#define HAVE_TDM 1
#endif
#endif

#ifdef HAVE_TDM
typedef __attribute__((ext_vector_type(4))) unsigned int u32x4;
typedef __attribute__((ext_vector_type(8))) int i32x8;
typedef __attribute__((ext_vector_type(4))) int i32x4;

__device__ __forceinline__ void tdm_load_2d_pad72(unsigned lds_addr, const void* gptr,
                                                  unsigned tensor_d0, unsigned tensor_d1,
                                                  unsigned stride0) {
  const unsigned long long ga = (unsigned long long)gptr;
  u32x4 g0;
  g0[0] = 1u;                                         // count=1, user descriptor
  g0[1] = lds_addr;                                   // LDS byte address
  g0[2] = (unsigned)(ga & 0xffffffffu);               // global_addr[31:0]
  g0[3] = (unsigned)((ga >> 32) & 0x01ffffffu)        // global_addr[56:32]
        | (2u << 30);                                 // type = 2 ("image")
  i32x8 g1;
  g1[0] = (1 << 16)                                   // data_size = 1 -> 2 bytes
        | (1 << 20)                                   // pad_enable
        | (4 << 22)                                   // pad_interval: 32 dwords
        | (3 << 25);                                  // pad_amount: 4 dwords
  g1[1] = (int)((tensor_d0 & 0xffffu) << 16);         // tensor_dim0[15:0] @ 63:48
  g1[2] = (int)(((tensor_d0 >> 16) & 0xffffu)         // tensor_dim0[31:16]
        | ((tensor_d1 & 0xffffu) << 16));             // tensor_dim1[15:0]
  g1[3] = (int)(((tensor_d1 >> 16) & 0xffffu)         // tensor_dim1[31:16]
        | (64u << 16));                               // tile_dim0 = 64
  g1[4] = 64;                                         // tile_dim1 = 64, tile_dim2 = 0
  g1[5] = (int)stride0;                               // tensor_dim0_stride[31:0]
  g1[6] = 0;                                          // stride0 hi, dim1_stride lo
  g1[7] = 0;
  i32x4 z4 = {0, 0, 0, 0};
#if defined(__clang_major__) && (__clang_major__ >= 23)
  i32x8 z8 = {0, 0, 0, 0, 0, 0, 0, 0};
  __builtin_amdgcn_tensor_load_to_lds(g0, g1, z4, z4, z8, 0);
#else
  __builtin_amdgcn_tensor_load_to_lds(g0, g1, z4, z4, 0);
#endif
}
#endif  // HAVE_TDM

// ---------------------------------------------------------------------------
// fp32 -> bf16 conversion (one-time; GEMMs run pure bf16 operands)
// ---------------------------------------------------------------------------
__global__ __launch_bounds__(256) void cvt_f32_bf16(const float* __restrict__ s,
                                                    bf16* __restrict__ d, size_t n) {
  size_t i = (size_t)blockIdx.x * 256 + threadIdx.x;
  size_t stride = (size_t)gridDim.x * 256;
  for (; i < n; i += stride) d[i] = (bf16)s[i];
}

// ---------------------------------------------------------------------------
// Tiled WMMA GEMM: C[M][N] = A[M][K] * Bw[N][K]^T  (both bf16 row-major).
// Block: 256 threads (8 waves), 128x128 output tile; each wave owns 32x64.
// Register-staged double buffering: tile i+1 is fetched from global into
// VGPRs while the WMMAs consume tile i from LDS; tile i+2 is prefetched
// into GL2 (global_prefetch_b8). LDS stride 72 -> conflict-free b128.
// ---------------------------------------------------------------------------
template <bool F32OUT>
__global__ __launch_bounds__(256) void gemm_wmma_bf16(const bf16* __restrict__ A,
                                                      const bf16* __restrict__ Bw,
                                                      void* __restrict__ Cout,
                                                      int N, int K) {
  __shared__ __align__(16) bf16 As[128 * 72];
  __shared__ __align__(16) bf16 Bs[128 * 72];

  const int m0 = blockIdx.y * 128;
  const int n0 = blockIdx.x * 128;
  const int t = threadIdx.x;
  const int wv = t >> 5;
  const int lane = t & 31;
  const int wm = (wv >> 1) * 32;   // wave's M offset inside tile (4 strips of 32)
  const int wn = (wv & 1) * 64;    // wave's N offset inside tile (2 strips of 64)
  const int lrow = t >> 1;         // cooperative tile-load row (0..127)
  const int lhalf = t & 1;         // which 32-element half of the row

  const bf16* ga = A + (size_t)(m0 + lrow) * K + lhalf * 32;
  const bf16* gb = Bw + (size_t)(n0 + lrow) * K + lhalf * 32;
  bf16* la = As + lrow * 72 + lhalf * 32;
  bf16* lb = Bs + lrow * 72 + lhalf * 32;

  v8f acc[2][4];
#pragma unroll
  for (int mi = 0; mi < 2; ++mi)
#pragma unroll
    for (int ni = 0; ni < 4; ++ni) acc[mi][ni] = v8f_zero();

  // stage tile 0 into registers
  v8bf ra[4], rb[4];
#pragma unroll
  for (int c = 0; c < 4; ++c) {
    ra[c] = *(const v8bf*)(ga + c * 8);
    rb[c] = *(const v8bf*)(gb + c * 8);
  }

  for (int k0 = 0; k0 < K; k0 += 64) {
    // commit staged tile to LDS
#pragma unroll
    for (int c = 0; c < 4; ++c) {
      *(v8bf*)(la + c * 8) = ra[c];
      *(v8bf*)(lb + c * 8) = rb[c];
    }
    __syncthreads();

    // fetch tile i+1 into registers (overlaps with WMMA below);
    // warm GL2 for tile i+2
    if (k0 + 64 < K) {
#pragma unroll
      for (int c = 0; c < 4; ++c) {
        ra[c] = *(const v8bf*)(ga + k0 + 64 + c * 8);
        rb[c] = *(const v8bf*)(gb + k0 + 64 + c * 8);
      }
      if (k0 + 128 < K) {
        __builtin_prefetch(ga + k0 + 128, 0, 1);
        __builtin_prefetch(gb + k0 + 128, 0, 1);
      }
    }

#pragma unroll
    for (int ks = 0; ks < 2; ++ks) {
      const int ko = ks * 32;
      v16bf a0 = frag_ld(As, wm + 0, 72, ko);
      v16bf a1 = frag_ld(As, wm + 16, 72, ko);
      v16bf b0 = frag_ld(Bs, wn + 0, 72, ko);
      v16bf b1 = frag_ld(Bs, wn + 16, 72, ko);
      v16bf b2 = frag_ld(Bs, wn + 32, 72, ko);
      v16bf b3 = frag_ld(Bs, wn + 48, 72, ko);
      acc[0][0] = WMMA_BF16(a0, b0, acc[0][0]);
      acc[0][1] = WMMA_BF16(a0, b1, acc[0][1]);
      acc[0][2] = WMMA_BF16(a0, b2, acc[0][2]);
      acc[0][3] = WMMA_BF16(a0, b3, acc[0][3]);
      acc[1][0] = WMMA_BF16(a1, b0, acc[1][0]);
      acc[1][1] = WMMA_BF16(a1, b1, acc[1][1]);
      acc[1][2] = WMMA_BF16(a1, b2, acc[1][2]);
      acc[1][3] = WMMA_BF16(a1, b3, acc[1][3]);
    }
    __syncthreads();
  }

  // Epilogue: C layout is VGPR r -> row r + 8*(lane>=16), lane&15 -> col.
  const int half = lane >> 4;
  const int nlo = lane & 15;
#pragma unroll
  for (int mi = 0; mi < 2; ++mi)
#pragma unroll
    for (int ni = 0; ni < 4; ++ni) {
      const int gn = n0 + wn + ni * 16 + nlo;
      const int gmb = m0 + wm + mi * 16 + 8 * half;
#pragma unroll
      for (int r = 0; r < 8; ++r) {
        const size_t idx = (size_t)(gmb + r) * N + gn;
        if (F32OUT) ((float*)Cout)[idx] = acc[mi][ni][r];
        else        ((bf16*)Cout)[idx] = (bf16)acc[mi][ni][r];
      }
    }
}

// ---------------------------------------------------------------------------
// RoPE + scatter into attention layouts.
// qkv:  [T][3072] bf16  (q: heads 0..31, k: heads 32..39, v: heads 40..47)
// Qr:   [B*H][S][64]    bf16 (WMMA A/B-friendly: row-major, K contiguous)
// Kr:   [B*HKV][S][64]  bf16
// Vt:   [B*HKV][64][S]  bf16 (pre-transposed so PV B-operand loads directly)
// cos/sin are the reference's [T,1,64] fp32 tables; cos[d]==cos[d+32].
// ---------------------------------------------------------------------------
__global__ __launch_bounds__(256) void rope_scatter(const bf16* __restrict__ qkv,
                                                    const float* __restrict__ cosb,
                                                    const float* __restrict__ sinb,
                                                    bf16* __restrict__ Qr,
                                                    bf16* __restrict__ Kr,
                                                    bf16* __restrict__ Vt) {
  const int tok = blockIdx.x;        // 0..4095
  const int b = tok >> 10;
  const int s = tok & 1023;
  for (int i = threadIdx.x; i < 48 * 32; i += 256) {
    const int hh = i >> 5;           // 0..47 combined head index
    const int d = i & 31;            // rotation pair index
    const float x0 = (float)qkv[(size_t)tok * 3072 + hh * 64 + d];
    const float x1 = (float)qkv[(size_t)tok * 3072 + hh * 64 + d + 32];
    if (hh < 40) {
      const float c = cosb[(size_t)tok * 64 + d];
      const float sn = sinb[(size_t)tok * 64 + d];
      const float y0 = x0 * c - x1 * sn;
      const float y1 = x1 * c + x0 * sn;
      if (hh < 32) {
        bf16* q = Qr + ((size_t)(b * 32 + hh) * 1024 + s) * 64;
        q[d] = (bf16)y0; q[d + 32] = (bf16)y1;
      } else {
        bf16* k = Kr + ((size_t)(b * 8 + (hh - 32)) * 1024 + s) * 64;
        k[d] = (bf16)y0; k[d + 32] = (bf16)y1;
      }
    } else {
      const int hk = hh - 40;
      Vt[((size_t)(b * 8 + hk) * 64 + d) * 1024 + s] = (bf16)x0;
      Vt[((size_t)(b * 8 + hk) * 64 + d + 32) * 1024 + s] = (bf16)x1;
    }
  }
}

// ---------------------------------------------------------------------------
// Causal GQA flash attention. Block = one (b,h) pair x 128 q rows (8 waves x 16).
// KV tiles of 64 keys; online softmax; QK^T and P*V through WMMA.
// K/V tiles arrive via TDM (tensor_load_to_lds + s_wait_tensorcnt) when
// available; hardware padding lands them directly in the stride-72 layout.
// ---------------------------------------------------------------------------
__global__ __launch_bounds__(256) void attn_flash(const bf16* __restrict__ Q,
                                                  const bf16* __restrict__ Kr,
                                                  const bf16* __restrict__ Vt,
                                                  bf16* __restrict__ Oa) {
  __shared__ __align__(16) bf16 Ks[64 * 72];
  __shared__ __align__(16) bf16 Vs[64 * 72];           // [dim][key]
  __shared__ __align__(16) bf16 Ps[8 * 16 * 72];       // per-wave P tile

  const int qb = blockIdx.x;       // q block (0..7), 128 rows each
  const int bh = blockIdx.y;       // 0..127
  const int b = bh >> 5, h = bh & 31, hk = h >> 2;
  const int t = threadIdx.x, wv = t >> 5, lane = t & 31;
  const int half = lane >> 4, nlo = lane & 15;

  // Q fragments for this wave's 16 rows (K dim = 64 -> two 32-wide fragments)
  const bf16* Qblk = Q + ((size_t)bh * 1024 + qb * 128 + wv * 16) * 64;
  const v16bf q0 = frag_ld(Qblk, 0, 64, 0);
  const v16bf q1 = frag_ld(Qblk, 0, 64, 32);

  v8f o[4];
#pragma unroll
  for (int dt = 0; dt < 4; ++dt) o[dt] = v8f_zero();
  float m_i[8], l_i[8];
#pragma unroll
  for (int r = 0; r < 8; ++r) { m_i[r] = -3.0e38f; l_i[r] = 0.0f; }

  const int row_s = qb * 128 + wv * 16 + 8 * half;     // + r -> this lane's q row
  const size_t kbase = (size_t)(b * 8 + hk) * 1024 * 64;
  const size_t vbase = (size_t)(b * 8 + hk) * 64 * 1024;
  bf16* pw = Ps + wv * (16 * 72);

  const int nkt = qb * 2 + 2;                          // causal: keys <= qb*128+127
  for (int kt = 0; kt < nkt; ++kt) {
    __syncthreads();                 // previous tile's LDS reads retired
#ifdef HAVE_TDM
    if (wv == 0) {
      // K tile: 64 rows of Kr[kt*64 ..][0..63], row stride 64 elements
      tdm_load_2d_pad72((unsigned)(size_t)(void*)Ks,
                        Kr + kbase + (size_t)kt * 64 * 64, 64u, 1024u, 64u);
      // V^T tile: 64 dim-rows of Vt[..][kt*64 .. kt*64+63], row stride 1024
      tdm_load_2d_pad72((unsigned)(size_t)(void*)Vs,
                        Vt + vbase + (size_t)kt * 64, 1024u, 64u, 1024u);
      __builtin_amdgcn_s_wait_tensorcnt(0);
    }
#else
    {   // cooperative K/V tile load: 64 rows x 64 elements each
      const int r = t >> 2, c = (t & 3) * 16;
      const bf16* gk = Kr + kbase + (size_t)(kt * 64 + r) * 64 + c;
      *(v8bf*)(Ks + r * 72 + c)     = *(const v8bf*)gk;
      *(v8bf*)(Ks + r * 72 + c + 8) = *(const v8bf*)(gk + 8);
      const bf16* gv = Vt + vbase + (size_t)r * 1024 + kt * 64 + c;
      *(v8bf*)(Vs + r * 72 + c)     = *(const v8bf*)gv;
      *(v8bf*)(Vs + r * 72 + c + 8) = *(const v8bf*)(gv + 8);
    }
#endif
    __syncthreads();

    // scores: 16 q rows x 64 keys = 4 WMMA tiles, 2 k-steps each
    v8f sc[4];
#pragma unroll
    for (int ni = 0; ni < 4; ++ni) {
      v16bf kf0 = frag_ld(Ks, ni * 16, 72, 0);
      v16bf kf1 = frag_ld(Ks, ni * 16, 72, 32);
      v8f z = v8f_zero();
      z = WMMA_BF16(q0, kf0, z);
      sc[ni] = WMMA_BF16(q1, kf1, z);
    }
    // scale + causal mask
#pragma unroll
    for (int ni = 0; ni < 4; ++ni) {
      const int col = kt * 64 + ni * 16 + nlo;
#pragma unroll
      for (int r = 0; r < 8; ++r) {
        const float sv = sc[ni][r] * 0.125f;           // HD^-0.5
        sc[ni][r] = (col <= row_s + r) ? sv : -3.0e38f;
      }
    }
    // online softmax update (row lives in a 16-lane half; xor 1/2/4/8 stays in-half)
    float mnew[8], alpha[8], rs[8];
#pragma unroll
    for (int r = 0; r < 8; ++r) {
      float mx = fmaxf(fmaxf(sc[0][r], sc[1][r]), fmaxf(sc[2][r], sc[3][r]));
      mx = fmaxf(mx, __shfl_xor(mx, 1, 32));
      mx = fmaxf(mx, __shfl_xor(mx, 2, 32));
      mx = fmaxf(mx, __shfl_xor(mx, 4, 32));
      mx = fmaxf(mx, __shfl_xor(mx, 8, 32));
      mnew[r] = fmaxf(m_i[r], mx);
      alpha[r] = __expf(m_i[r] - mnew[r]);
      m_i[r] = mnew[r];
      rs[r] = 0.0f;
    }
#pragma unroll
    for (int ni = 0; ni < 4; ++ni)
#pragma unroll
      for (int r = 0; r < 8; ++r) {
        const float p = __expf(sc[ni][r] - mnew[r]);
        rs[r] += p;
        pw[(r + 8 * half) * 72 + ni * 16 + nlo] = (bf16)p;  // C-layout -> LDS
      }
#pragma unroll
    for (int r = 0; r < 8; ++r) {
      float v = rs[r];
      v += __shfl_xor(v, 1, 32);
      v += __shfl_xor(v, 2, 32);
      v += __shfl_xor(v, 4, 32);
      v += __shfl_xor(v, 8, 32);
      l_i[r] = l_i[r] * alpha[r] + v;
    }
#pragma unroll
    for (int dt = 0; dt < 4; ++dt)
#pragma unroll
      for (int r = 0; r < 8; ++r) o[dt][r] *= alpha[r];

    __syncthreads();   // make P visible in A-layout reads; all waves in lockstep

    // O += P * V  (A = P from LDS, B = V^T tile: lane = dim col, K = key)
    const v16bf p0 = frag_ld(pw, 0, 72, 0);
    const v16bf p1 = frag_ld(pw, 0, 72, 32);
#pragma unroll
    for (int dt = 0; dt < 4; ++dt) {
      v16bf vf0 = frag_ld(Vs, dt * 16, 72, 0);
      v16bf vf1 = frag_ld(Vs, dt * 16, 72, 32);
      o[dt] = WMMA_BF16(p0, vf0, o[dt]);
      o[dt] = WMMA_BF16(p1, vf1, o[dt]);
    }
  }

  // normalize + store to [T][2048] bf16 (token-major, heads concatenated)
#pragma unroll
  for (int r = 0; r < 8; ++r) {
    const float inv = 1.0f / l_i[r];
    const size_t tok = (size_t)b * 1024 + row_s + r;
#pragma unroll
    for (int dt = 0; dt < 4; ++dt)
      Oa[tok * 2048 + h * 64 + dt * 16 + nlo] = (bf16)(o[dt][r] * inv);
  }
}

// ---------------------------------------------------------------------------
// Host-side orchestration
// ---------------------------------------------------------------------------
extern "C" void kernel_launch(void* const* d_in, const int* in_sizes, int n_in,
                              void* d_out, int out_size, void* d_ws, size_t ws_size,
                              hipStream_t stream) {
  (void)in_sizes; (void)n_in; (void)out_size; (void)ws_size;
  constexpr int T = 4096, DM = 2048, NQKV = 3072;
  constexpr size_t N_HID = (size_t)T * DM;       // 8.4M
  constexpr size_t N_WIN = (size_t)NQKV * DM;    // 6.3M
  constexpr size_t N_WOUT = (size_t)DM * DM;     // 4.2M
  constexpr size_t N_QKV = (size_t)T * NQKV;     // 12.6M
  constexpr size_t N_Q = (size_t)T * 2048;       // 8.4M
  constexpr size_t N_KV = (size_t)T * 512;       // 2.1M

  const float* hs   = (const float*)d_in[0];
  const float* cosb = (const float*)d_in[1];
  const float* sinb = (const float*)d_in[2];
  const float* w_in = (const float*)d_in[3];
  const float* w_out = (const float*)d_in[4];
  // d_in[5] = position_ids: unused (cos/sin tables already expanded)

  char* ws = (char*)d_ws;
  size_t off = 0;
  auto carve = [&](size_t elems) {
    bf16* p = (bf16*)(ws + off);
    off = (off + elems * sizeof(bf16) + 255) & ~(size_t)255;
    return p;
  };
  bf16* hid_bf  = carve(N_HID);    // also reused as attn_out after GEMM1 retires
  bf16* win_bf  = carve(N_WIN);
  bf16* wout_bf = carve(N_WOUT);
  bf16* qkv_bf  = carve(N_QKV);
  bf16* Qr      = carve(N_Q);
  bf16* Kr      = carve(N_KV);
  bf16* Vt      = carve(N_KV);
  bf16* attn_bf = hid_bf;          // alias: hidden_bf16 is dead after GEMM1

  // 1) fp32 -> bf16 operand conversion
  cvt_f32_bf16<<<(int)((N_HID + 255) / 256), 256, 0, stream>>>(hs, hid_bf, N_HID);
  cvt_f32_bf16<<<(int)((N_WIN + 255) / 256), 256, 0, stream>>>(w_in, win_bf, N_WIN);
  cvt_f32_bf16<<<(int)((N_WOUT + 255) / 256), 256, 0, stream>>>(w_out, wout_bf, N_WOUT);

  // 2) QKV projection: qkv[T][3072] = hidden[T][2048] * w_in[3072][2048]^T
  gemm_wmma_bf16<false><<<dim3(NQKV / 128, T / 128), 256, 0, stream>>>(
      hid_bf, win_bf, (void*)qkv_bf, NQKV, DM);

  // 3) RoPE + scatter into attention layouts
  rope_scatter<<<T, 256, 0, stream>>>(qkv_bf, cosb, sinb, Qr, Kr, Vt);

  // 4) causal GQA flash attention -> attn_bf [T][2048]
  attn_flash<<<dim3(8, 128), 256, 0, stream>>>(Qr, Kr, Vt, attn_bf);

  // 5) output projection (fp32 result straight to d_out)
  gemm_wmma_bf16<true><<<dim3(DM / 128, T / 128), 256, 0, stream>>>(
      attn_bf, wout_bf, d_out, DM, DM);
}
